// WFPN_51951924412997
// MI455X (gfx1250) — matrix operand order repaired and checked by reading, not compiled, not run on hardware
//
#include <hip/hip_runtime.h>
#include <hip/hip_bf16.h>

// WS requirement: ~50 MB (ori_fe + theta/phi/g bf16 + per-batch score matrix +
// bsf + per-level scratch). All launches on `stream`, no allocation here.

#define PI_HALF 1.57079632679489662f
#define EPS_    1e-7f

typedef __bf16 v16bf __attribute__((ext_vector_type(16)));
typedef float  v8f   __attribute__((ext_vector_type(8)));

constexpr int kB   = 4;
constexpr int kC   = 256;
constexpr int kHW  = 2500;   // 50*50
constexpr int kHWP = 2528;   // padded to 79*32 (multiple of 32 and 16)
constexpr int kGH  = 50;

__device__ __forceinline__ __bf16 f2b(float f) { return (__bf16)f; }
__device__ __forceinline__ float  b2f(__bf16 b) { return (float)b; }

__device__ __forceinline__ v8f wmma_bf16(v16bf a, v16bf b, v8f c) {
  return __builtin_amdgcn_wmma_f32_16x16x32_bf16(false, a, false, b, (short)0, c,
                                                 false, false);
}

// A-matrix 16x32 bf16 fragment, contiguous K order (same as B):
// lane[0..15]: m=lane, K = e; lane[16..31]: m=lane-16, K = 16 + e.
// Per-lane K runs are contiguous-16 -> lowers to 2x ds_load_b128.
__device__ __forceinline__ v16bf load_a_frag(const __bf16* s, int lane, int stride) {
  v16bf a;
  int m  = lane & 15;
  int h8 = (lane >> 4) << 4;
#pragma unroll
  for (int e = 0; e < 16; ++e)
    a[e] = s[m * stride + (e + h8)];
  return a;
}

// B-matrix 32x16 bf16 fragment: n = lane%16, K = e + 16*(lane/16).
// Swizzled staging: tile stored in fragment order so each lane reads 16
// contiguous bf16 (2x ds_load_b128). Store index for element (K,n):
__device__ __forceinline__ int bsw_idx(int K, int n) {
  return (n + ((K >> 4) << 4)) * 16 + (K & 15);
}
__device__ __forceinline__ v16bf load_b_sw(const __bf16* s, int lane) {
  v16bf b;
  const __bf16* base = s + lane * 16;
#pragma unroll
  for (int e = 0; e < 16; ++e) b[e] = base[e];
  return b;
}

// Monotone float <-> u32 encoding for atomic min/max.
__device__ __forceinline__ unsigned fenc(float f) {
  unsigned u = __float_as_uint(f);
  return (u & 0x80000000u) ? ~u : (u | 0x80000000u);
}
__device__ __forceinline__ float fdec(unsigned u) {
  unsigned b = (u & 0x80000000u) ? (u & 0x7FFFFFFFu) : ~u;
  return __uint_as_float(b);
}

// ---------------------------------------------------------------------------
// 1. Fuse pyramid levels to ori_fe (b,256,50,50).
// ---------------------------------------------------------------------------
__global__ void fuse_kernel(const float* __restrict__ x0, const float* __restrict__ x1,
                            const float* __restrict__ x2, const float* __restrict__ x3,
                            const float* __restrict__ x4, float* __restrict__ fe) {
  int idx = blockIdx.x * blockDim.x + threadIdx.x;
  if (idx >= kB * kC * kHW) return;
  int p  = idx % kHW;
  int bc = idx / kHW;
  int y = p / kGH, xx = p % kGH;

  const float* p0 = x0 + (size_t)bc * 200 * 200;
  float m0 = -3.0e38f;
#pragma unroll
  for (int dy = 0; dy < 4; ++dy)
#pragma unroll
    for (int dx = 0; dx < 4; ++dx)
      m0 = fmaxf(m0, p0[(4 * y + dy) * 200 + 4 * xx + dx]);

  const float* p1 = x1 + (size_t)bc * 100 * 100;
  float m1 = -3.0e38f;
#pragma unroll
  for (int dy = 0; dy < 2; ++dy)
#pragma unroll
    for (int dx = 0; dx < 2; ++dx)
      m1 = fmaxf(m1, p1[(2 * y + dy) * 100 + 2 * xx + dx]);

  float m2 = x2[(size_t)bc * kHW + p];
  float m3 = x3[(size_t)bc * 625 + (y / 2) * 25 + (xx / 2)];
  int ry = (y * 13) / kGH, rx = (xx * 13) / kGH;
  float m4 = x4[(size_t)bc * 169 + ry * 13 + rx];

  fe[idx] = (m0 + m1 + m2 + m3 + m4) * 0.2f;
}

// ---------------------------------------------------------------------------
// 2. 1x1 conv as WMMA GEMM: out[m=c_out][n=pixel] = W(256x256) @ X(256xHW)+b.
//    transposed=1 -> store [b][p][c] (theta/g), else [b][c][p] (phi). bf16 out.
// ---------------------------------------------------------------------------
__global__ void __launch_bounds__(128)
conv1x1_wmma(const float* __restrict__ X, const float* __restrict__ Wt,
             const float* __restrict__ bias, __bf16* __restrict__ out, int transposed) {
  __shared__ __bf16 Ws[256 * 32];  // weight K-chunk, all 256 m rows (A layout: row-major)
  __shared__ __bf16 Xs[32 * 16];   // activation K-chunk x 16 pixels, fragment-swizzled
  int b = blockIdx.y;
  int p0 = blockIdx.x * 16;
  int tid = threadIdx.x, lane = tid & 31, wv = tid >> 5;

  v8f acc[4];
#pragma unroll
  for (int i = 0; i < 4; ++i) acc[i] = (v8f){0.f, 0.f, 0.f, 0.f, 0.f, 0.f, 0.f, 0.f};

  for (int k0 = 0; k0 < kC; k0 += 32) {
    for (int e = tid; e < 256 * 32; e += 128) {
      int m = e >> 5, kk = e & 31;
      Ws[e] = f2b(Wt[m * kC + k0 + kk]);
    }
    for (int e = tid; e < 32 * 16; e += 128) {
      int kk = e >> 4, n = e & 15, p = p0 + n;
      Xs[bsw_idx(kk, n)] =
          (p < kHW) ? f2b(X[((size_t)(b * kC) + (k0 + kk)) * kHW + p]) : f2b(0.f);
    }
    __syncthreads();
    v16bf bfrag = load_b_sw(Xs, lane);
#pragma unroll
    for (int mt = 0; mt < 4; ++mt) {
      int m0 = (wv * 4 + mt) * 16;
      v16bf afrag = load_a_frag(Ws + m0 * 32, lane, 32);
      acc[mt] = wmma_bf16(afrag, bfrag, acc[mt]);
    }
    __syncthreads();
  }
#pragma unroll
  for (int mt = 0; mt < 4; ++mt) {
    int m0 = (wv * 4 + mt) * 16;
#pragma unroll
    for (int r = 0; r < 8; ++r) {
      int m = m0 + r + ((lane >> 4) << 3);  // C/D: M = r + 8*(lane/16)
      int p = p0 + (lane & 15);             // C/D: N = lane%16
      float v = (p < kHW) ? (acc[mt][r] + bias[m]) : 0.f;
      if (transposed)
        out[((size_t)b * kHWP + p) * kC + m] = f2b(v);
      else
        out[((size_t)(b * kC) + m) * kHWP + p] = f2b(v);
    }
  }
}

// ---------------------------------------------------------------------------
// 3. Attention scores S[i][j] = theta_t(i,:) . phi(:,j). One 16x16 tile per
//    single-wave workgroup, K=256 in 8 chunks. bf16 output (stays in L2).
// ---------------------------------------------------------------------------
__global__ void __launch_bounds__(32)
scores_wmma(const __bf16* __restrict__ th, const __bf16* __restrict__ ph,
            __bf16* __restrict__ S, int b) {
  __shared__ __bf16 As[16 * 32];
  __shared__ __bf16 Bs[32 * 16];  // fragment-swizzled
  int i0 = blockIdx.x * 16, j0 = blockIdx.y * 16;
  int lane = threadIdx.x & 31;
  v8f acc = (v8f){0.f, 0.f, 0.f, 0.f, 0.f, 0.f, 0.f, 0.f};
  const __bf16* thb = th + (size_t)b * kHWP * kC;
  const __bf16* phb = ph + (size_t)b * kC * kHWP;

  for (int k0 = 0; k0 < kC; k0 += 32) {
    for (int e = lane; e < 512; e += 32) {
      int m = e >> 5, kk = e & 31;
      As[e] = thb[(size_t)(i0 + m) * kC + k0 + kk];
    }
    for (int e = lane; e < 512; e += 32) {
      int kk = e >> 4, n = e & 15;
      Bs[bsw_idx(kk, n)] = phb[(size_t)(k0 + kk) * kHWP + j0 + n];
    }
    __syncthreads();
    acc = wmma_bf16(load_a_frag(As, lane, 32), load_b_sw(Bs, lane), acc);
    __syncthreads();
  }
#pragma unroll
  for (int r = 0; r < 8; ++r) {
    int i = i0 + r + ((lane >> 4) << 3);
    int j = j0 + (lane & 15);
    S[(size_t)i * kHWP + j] = f2b(acc[r]);
  }
}

// ---------------------------------------------------------------------------
// 4. Row softmax over j<2500, in place (bf16). Pad rows/cols -> 0.
// ---------------------------------------------------------------------------
__global__ void __launch_bounds__(128) softmax_kernel(__bf16* __restrict__ S) {
  int i = blockIdx.x, tid = threadIdx.x;
  __shared__ float red[128];
  __bf16* row = S + (size_t)i * kHWP;
  if (i >= kHW) {
    for (int j = tid; j < kHWP; j += 128) row[j] = f2b(0.f);
    return;
  }
  float mx = -3.0e38f;
  for (int j = tid; j < kHW; j += 128) mx = fmaxf(mx, b2f(row[j]));
  red[tid] = mx;
  __syncthreads();
  for (int st = 64; st > 0; st >>= 1) {
    if (tid < st) red[tid] = fmaxf(red[tid], red[tid + st]);
    __syncthreads();
  }
  mx = red[0];
  __syncthreads();
  float sm = 0.f;
  for (int j = tid; j < kHW; j += 128) sm += __expf(b2f(row[j]) - mx);
  red[tid] = sm;
  __syncthreads();
  for (int st = 64; st > 0; st >>= 1) {
    if (tid < st) red[tid] += red[tid + st];
    __syncthreads();
  }
  float inv = 1.0f / red[0];
  __syncthreads();
  for (int j = tid; j < kHW; j += 128) row[j] = f2b(__expf(b2f(row[j]) - mx) * inv);
  for (int j = kHW + tid; j < kHWP; j += 128) row[j] = f2b(0.f);
}

// ---------------------------------------------------------------------------
// 5. y = P @ g_t (16 rows x 256 c), then fused out-proj:
//    bsf = ori_fe + out_w @ y + out_b. Two WMMA GEMMs per block.
//    All B operands staged in fragment order -> ds_load_b128 only.
// ---------------------------------------------------------------------------
__global__ void __launch_bounds__(128)
pv_out_wmma(const __bf16* __restrict__ P, const __bf16* __restrict__ G,
            const float* __restrict__ ow, const float* __restrict__ ob,
            const float* __restrict__ fe, float* __restrict__ bsf, int b) {
  __shared__ __bf16 scratch[17 * 512];  // phase1: Ps(512)+Gs(4*2048); phase2: Wo(8192)
  __shared__ __bf16 yfr[8 * 512];       // y in fragment order per 32-wide K chunk
  int tid = threadIdx.x, lane = tid & 31, wv = tid >> 5;
  int i0 = blockIdx.x * 16;

  __bf16* Ps = scratch;        // A tile [16 rows][32 j]
  __bf16* Gs = scratch + 512;  // per-wave 4 fragment-ordered B tiles (4*512 each)

  v8f acc[4];
#pragma unroll
  for (int i = 0; i < 4; ++i) acc[i] = (v8f){0.f, 0.f, 0.f, 0.f, 0.f, 0.f, 0.f, 0.f};

  const __bf16* Gb = G + (size_t)b * kHWP * kC;

  for (int kt = 0; kt < kHWP / 32; ++kt) {  // 79 K chunks over j
    int k0 = kt * 32;
    for (int e = tid; e < 512; e += 128) {
      int m = e >> 5, kk = e & 31;
      Ps[e] = P[(size_t)(i0 + m) * kHWP + k0 + kk];
    }
    {  // wave stages its 64-wide c slab: b128 global loads, scatter to frag order
      const __bf16* src = Gb + (size_t)(k0 + lane) * kC + wv * 64;
      __bf16 tmp[64];
      const uint4* s4 = reinterpret_cast<const uint4*>(src);
      uint4* t4 = reinterpret_cast<uint4*>(tmp);
#pragma unroll
      for (int q = 0; q < 8; ++q) t4[q] = s4[q];
      __bf16* g = Gs + wv * 2048;
      int e = lane & 15;
      int lh = (lane >> 4) << 4;
#pragma unroll
      for (int cc = 0; cc < 64; ++cc) {
        int ct = cc >> 4, n = cc & 15;
        g[ct * 512 + (n + lh) * 16 + e] = tmp[cc];
      }
    }
    __syncthreads();
    v16bf afrag = load_a_frag(Ps, lane, 32);
#pragma unroll
    for (int ct = 0; ct < 4; ++ct)
      acc[ct] = wmma_bf16(afrag, load_b_sw(Gs + wv * 2048 + ct * 512, lane), acc[ct]);
    __syncthreads();
  }
  // y -> LDS in fragment order (B operand of phase 2)
#pragma unroll
  for (int ct = 0; ct < 4; ++ct) {
    int c0 = wv * 64 + ct * 16;
    int cl = lane & 15;
#pragma unroll
    for (int r = 0; r < 8; ++r) {
      int c = c0 + cl;
      int p = r + ((lane >> 4) << 3);
      int Kp = c & 31;
      yfr[(c >> 5) * 512 + (p + ((Kp >> 4) << 4)) * 16 + (Kp & 15)] = f2b(acc[ct][r]);
    }
  }
  __syncthreads();

  // phase 2: out-projection GEMM M=c_out(256) K=c(256) N=16 pixels
  __bf16* Wo = scratch;
  v8f acc2[4];
#pragma unroll
  for (int i = 0; i < 4; ++i) acc2[i] = (v8f){0.f, 0.f, 0.f, 0.f, 0.f, 0.f, 0.f, 0.f};

  for (int k0 = 0; k0 < kC; k0 += 32) {
    for (int e = tid; e < 256 * 32; e += 128) {
      int m = e >> 5, kk = e & 31;
      Wo[e] = f2b(ow[m * kC + k0 + kk]);
    }
    __syncthreads();
    v16bf bfrag = load_b_sw(yfr + (k0 >> 5) * 512, lane);
#pragma unroll
    for (int mt = 0; mt < 4; ++mt) {
      int m0 = (wv * 4 + mt) * 16;
      v16bf afrag = load_a_frag(Wo + m0 * 32, lane, 32);
      acc2[mt] = wmma_bf16(afrag, bfrag, acc2[mt]);
    }
    __syncthreads();
  }
#pragma unroll
  for (int mt = 0; mt < 4; ++mt) {
    int m0 = (wv * 4 + mt) * 16;
#pragma unroll
    for (int r = 0; r < 8; ++r) {
      int co = m0 + r + ((lane >> 4) << 3);
      int p = i0 + (lane & 15);
      if (p < kHW) {
        size_t o = ((size_t)(b * kC) + co) * kHW + p;
        bsf[o] = acc2[mt][r] + ob[co] + fe[o];
      }
    }
  }
}

// ---------------------------------------------------------------------------
// Per-level stat kernels
// stats layout (u32): [0:4) bm_min, [4:8) bm_max, [8:12) r_min, [12:16) r_max,
//                     [16:20) c_min, [20:24) c_max
// ---------------------------------------------------------------------------
__global__ void init_stats_kernel(unsigned* s) {
  int t = threadIdx.x;
  if (t < 24) s[t] = (((t >> 2) & 1) == 0) ? 0xFFFFFFFFu : 0u;
}

__global__ void __launch_bounds__(128)
rowcol_kernel(const float* __restrict__ x, float* __restrict__ rowm,
              float* __restrict__ colm, unsigned* __restrict__ stats, int H, int W) {
  int b = blockIdx.y, idx = blockIdx.x, tid = threadIdx.x;
  __shared__ float red[128];
  const float* base = x + (size_t)b * kC * H * W;  // channel 0
  float s = 0.f;
  if (idx < H) {
    for (int w = tid; w < W; w += 128) s += base[idx * W + w];
  } else {
    int j = idx - H;
    for (int h = tid; h < H; h += 128) s += base[h * W + j];
  }
  red[tid] = s;
  __syncthreads();
  for (int st = 64; st > 0; st >>= 1) {
    if (tid < st) red[tid] += red[tid + st];
    __syncthreads();
  }
  if (tid == 0) {
    if (idx < H) {
      float m = red[0] / (float)W;
      rowm[b * H + idx] = m;
      atomicMin(&stats[8 + b], fenc(m));
      atomicMax(&stats[12 + b], fenc(m));
    } else {
      float m = red[0] / (float)H;
      colm[b * W + (idx - H)] = m;
      atomicMin(&stats[16 + b], fenc(m));
      atomicMax(&stats[20 + b], fenc(m));
    }
  }
}

__global__ void __launch_bounds__(256)
conv3x3_kernel(const float* __restrict__ x, const float* __restrict__ w9,
               const float* __restrict__ bptr, float* __restrict__ bm,
               unsigned* __restrict__ stats, int H, int W) {
  __shared__ float ws[kC * 9];
  __shared__ float rmin[256], rmax[256];
  int b = blockIdx.y, tid = threadIdx.x;
  for (int i = tid; i < kC * 9; i += 256) ws[i] = w9[i];
  __syncthreads();
  int pix = blockIdx.x * 256 + tid;
  float v = 0.f;
  bool valid = pix < H * W;
  if (valid) {
    int h = pix / W, w = pix % W;
    float acc = bptr[0];
    const float* xb = x + (size_t)b * kC * H * W;
    for (int c = 0; c < kC; ++c) {
      const float* xc = xb + (size_t)c * H * W;
      const float* wc = ws + c * 9;
#pragma unroll
      for (int ky = 0; ky < 3; ++ky) {
        int hh = h + ky - 1;
        if (hh < 0 || hh >= H) continue;
#pragma unroll
        for (int kx = 0; kx < 3; ++kx) {
          int wi = w + kx - 1;
          if (wi < 0 || wi >= W) continue;
          acc += xc[hh * W + wi] * wc[ky * 3 + kx];
        }
      }
    }
    v = fmaxf(acc, 0.f);
    bm[(size_t)b * H * W + pix] = v;
  }
  rmin[tid] = valid ? v : 3.0e38f;
  rmax[tid] = valid ? v : -3.0e38f;
  __syncthreads();
  for (int st = 128; st > 0; st >>= 1) {
    if (tid < st) {
      rmin[tid] = fminf(rmin[tid], rmin[tid + st]);
      rmax[tid] = fmaxf(rmax[tid], rmax[tid + st]);
    }
    __syncthreads();
  }
  if (tid == 0) {
    atomicMin(&stats[0 + b], fenc(rmin[0]));
    atomicMax(&stats[4 + b], fenc(rmax[0]));
  }
}

__global__ void __launch_bounds__(256)
final_kernel(const float* __restrict__ xin, const float* __restrict__ bsf,
             const float* __restrict__ bm, const float* __restrict__ rowm,
             const float* __restrict__ colm, const unsigned* __restrict__ stats,
             float* __restrict__ out, int H, int W, long long n) {
  long long idx = (long long)blockIdx.x * 256 + threadIdx.x;
  if (idx >= n) return;
  if (idx + 8192 < n) __builtin_prefetch(&xin[idx + 8192], 0, 0);  // global_prefetch_b8
  int w = (int)(idx % W);
  long long t = idx / W;
  int h = (int)(t % H);
  t /= H;
  int c = (int)(t % kC);
  int b = (int)(t / kC);

  float bmn = fdec(stats[0 + b]), bmx = fdec(stats[4 + b]);
  float rmn = fdec(stats[8 + b]), rmx = fdec(stats[12 + b]);
  float cmn = fdec(stats[16 + b]), cmx = fdec(stats[20 + b]);
  float p1 = rmn * cmn, p2 = rmn * cmx, p3 = rmx * cmn, p4 = rmx * cmx;
  float amn = fminf(fminf(p1, p2), fminf(p3, p4));
  float amx = fmaxf(fmaxf(p1, p2), fmaxf(p3, p4));

  float avg = (rowm[b * H + h] * colm[b * W + w] - amn) / (amx - amn + EPS_);
  float breg = (bm[(size_t)b * H * W + (size_t)h * W + w] - bmn) / (bmx - bmn + EPS_);
  int sh = (h * kGH) / H, sw = (w * kGH) / W;
  float bs = bsf[((size_t)(b * kC) + c) * kHW + sh * kGH + sw];
  float att = fmaxf(bs * cosf((avg - breg) * PI_HALF), 0.f);
  out[idx] = xin[idx] + att;
}

// ---------------------------------------------------------------------------
extern "C" void kernel_launch(void* const* d_in, const int* in_sizes, int n_in,
                              void* d_out, int out_size, void* d_ws, size_t ws_size,
                              hipStream_t stream) {
  (void)in_sizes; (void)n_in; (void)out_size; (void)ws_size;
  const float* x[5];
  for (int i = 0; i < 5; ++i) x[i] = (const float*)d_in[i];
  const float* rc_w = (const float*)d_in[5];
  const float* rc_b = (const float*)d_in[6];
  const float* g_w  = (const float*)d_in[7];
  const float* g_b  = (const float*)d_in[8];
  const float* th_w = (const float*)d_in[9];
  const float* th_b = (const float*)d_in[10];
  const float* ph_w = (const float*)d_in[11];
  const float* ph_b = (const float*)d_in[12];
  const float* o_w  = (const float*)d_in[13];
  const float* o_b  = (const float*)d_in[14];

  char* base = (char*)d_ws;
  size_t off = 0;
  auto carve = [&](size_t bytes) -> char* {
    char* p = base + off;
    off = (off + bytes + 255) & ~(size_t)255;
    return p;
  };
  float*    fe   = (float*)carve((size_t)kB * kC * kHW * 4);
  __bf16*   tht  = (__bf16*)carve((size_t)kB * kHWP * kC * 2);  // [b][p][c]
  __bf16*   phm  = (__bf16*)carve((size_t)kB * kC * kHWP * 2);  // [b][c][p]
  __bf16*   gt   = (__bf16*)carve((size_t)kB * kHWP * kC * 2);  // [b][p][c]
  __bf16*   S    = (__bf16*)carve((size_t)kHWP * kHWP * 2);     // per-batch reuse
  float*    bsf  = (float*)carve((size_t)kB * kC * kHW * 4);
  float*    bm   = (float*)carve((size_t)kB * 200 * 200 * 4);
  float*    rowm = (float*)carve((size_t)kB * 200 * 4);
  float*    colm = (float*)carve((size_t)kB * 200 * 4);
  unsigned* stats = (unsigned*)carve(24 * 4);

  // 1) fuse pyramid -> ori_fe
  fuse_kernel<<<(kB * kC * kHW + 255) / 256, 256, 0, stream>>>(x[0], x[1], x[2], x[3],
                                                               x[4], fe);
  // 2) theta / phi / g 1x1 convs (WMMA bf16)
  dim3 g1(kHWP / 16, kB);
  conv1x1_wmma<<<g1, 128, 0, stream>>>(fe, th_w, th_b, tht, 1);
  conv1x1_wmma<<<g1, 128, 0, stream>>>(fe, ph_w, ph_b, phm, 0);
  conv1x1_wmma<<<g1, 128, 0, stream>>>(fe, g_w, g_b, gt, 1);

  // 3-5) attention per batch (score matrix stays in 192MB L2)
  for (int b = 0; b < kB; ++b) {
    scores_wmma<<<dim3(157, kHWP / 16), 32, 0, stream>>>(tht, phm, S, b);
    softmax_kernel<<<157 * 16, 128, 0, stream>>>(S);
    pv_out_wmma<<<157, 128, 0, stream>>>(S, gt, o_w, o_b, fe, bsf, b);
  }

  // 6) per-level attention re-weighting
  const int Hs[5] = {200, 100, 50, 25, 13};
  float* out = (float*)d_out;
  size_t ooff = 0;
  for (int i = 0; i < 5; ++i) {
    int H = Hs[i], W = Hs[i];
    init_stats_kernel<<<1, 32, 0, stream>>>(stats);
    rowcol_kernel<<<dim3(H + W, kB), 128, 0, stream>>>(x[i], rowm, colm, stats, H, W);
    conv3x3_kernel<<<dim3((H * W + 255) / 256, kB), 256, 0, stream>>>(
        x[i], rc_w + (size_t)i * kC * 9, rc_b + i, bm, stats, H, W);
    long long n = (long long)kB * kC * H * W;
    final_kernel<<<(unsigned)((n + 255) / 256), 256, 0, stream>>>(
        x[i], bsf, bm, rowm, colm, stats, out + ooff, H, W, n);
    ooff += (size_t)n;
  }
}
